// MultiDiscriminator_77034533421573
// MI455X (gfx1250) — matrix operand order repaired and checked by reading, not compiled, not run on hardware
//
#include <hip/hip_runtime.h>
#include <hip/hip_bf16.h>
#include <math.h>

typedef __attribute__((ext_vector_type(16))) _Float16 v16h;
typedef __attribute__((ext_vector_type(8)))  _Float16 v8h;
typedef __attribute__((ext_vector_type(8)))  float    v8f;

#define B_TOK   16384
#define OBS     384
#define ACTN    128
#define DIN     512
#define HDIM    512
#define NE      8
#define SEG_PAD 64
#define MPAD    16896   // 16384 + 8*64, multiple of 64

// ---- workspace layout (bytes, all 256B aligned) ----
#define WS_OFFS    0                               // 9 ints: padded segment offsets
#define WS_CURSOR  256                             // 8 ints: scatter cursors
#define WS_ROWTOK  512                             // MPAD ints: row -> token (-1 = pad)
#define WS_W1T     68096                           // [E][H][DIN] f16 (transposed)
#define WS_W2T     (WS_W1T + NE*HDIM*DIN*2)        // [E][H][H]   f16 (transposed)
#define WS_X       (WS_W2T + NE*HDIM*HDIM*2)       // [MPAD][DIN] f16 (gathered input)
#define WS_H1      (WS_X + (size_t)MPAD*DIN*2)     // [MPAD][H]   f16
// H2 aliases X (X is dead after GEMM1)

// ---------------- routing ----------------
__global__ void k_route_count(const int* __restrict__ idx, int* __restrict__ offs,
                              int* __restrict__ cursor, int* __restrict__ rowToken) {
    __shared__ int cnt[NE];
    const int tid = threadIdx.x;
    if (tid < NE) cnt[tid] = 0;
    __syncthreads();
    for (int t = tid; t < B_TOK; t += blockDim.x)
        atomicAdd(&cnt[idx[t] & 7], 1);
    for (int r = tid; r < MPAD; r += blockDim.x)
        rowToken[r] = -1;
    __syncthreads();
    if (tid == 0) {
        int acc = 0;
        offs[0] = 0;
        for (int e = 0; e < NE; ++e) {
            cursor[e] = acc;
            acc += (cnt[e] + SEG_PAD - 1) / SEG_PAD * SEG_PAD;
            offs[e + 1] = acc;
        }
    }
}

__global__ void k_route_scatter(const int* __restrict__ idx, int* __restrict__ cursor,
                                int* __restrict__ rowToken) {
    const int t = blockIdx.x * blockDim.x + threadIdx.x;
    if (t >= B_TOK) return;
    const int e = idx[t] & 7;
    const int pos = atomicAdd(&cursor[e], 1);
    rowToken[pos] = t;   // order within expert is arbitrary; per-token result is order-independent
}

// ---------------- data conversion ----------------
__global__ void k_build_x(const float* __restrict__ obs, const float* __restrict__ act,
                          const int* __restrict__ rowToken, _Float16* __restrict__ X) {
    const long i = (long)blockIdx.x * blockDim.x + threadIdx.x;   // MPAD*DIN threads
    const int r = (int)(i >> 9);
    const int c = (int)(i & 511);
    const int t = rowToken[r];
    float v = 0.f;
    if (t >= 0) v = (c < OBS) ? obs[(long)t * OBS + c] : act[(long)t * ACTN + (c - OBS)];
    X[i] = (_Float16)v;
}

// W1t[e][h][d] = W1[e][d][h] ; W2t[e][h][k] = W2[e][k][h]  (f32 -> f16)
__global__ void k_transpose_w(const float* __restrict__ W1, const float* __restrict__ W2,
                              _Float16* __restrict__ W1t, _Float16* __restrict__ W2t) {
    const long per = (long)NE * HDIM * DIN;
    const long i = (long)blockIdx.x * blockDim.x + threadIdx.x;   // 2*per threads
    if (i < per) {
        const int e = (int)(i / (HDIM * DIN));
        const int rem = (int)(i % (HDIM * DIN));
        const int h = rem / DIN, d = rem % DIN;
        W1t[i] = (_Float16)W1[((long)e * DIN + d) * HDIM + h];
    } else {
        const long j = i - per;
        const int e = (int)(j / (HDIM * HDIM));
        const int rem = (int)(j % (HDIM * HDIM));
        const int h = rem / HDIM, k = rem % HDIM;
        W2t[j] = (_Float16)W2[((long)e * HDIM + k) * HDIM + h];
    }
}

// ---------------- WMMA GEMM + bias + ReLU ----------------
// A: [MPAD][K] f16 row-major, Wt: [E][512][K] f16 row-major (N rows), Out: [MPAD][512] f16
// Block = 8 waves; block tile = 64(M) x 128(N); wave tile = 64(M) x 16(N).
__global__ __launch_bounds__(256) void k_gemm_relu(
    const _Float16* __restrict__ A, const _Float16* __restrict__ Wt,
    const float* __restrict__ bias, const int* __restrict__ offs,
    _Float16* __restrict__ Out, int K) {
    const int lane  = threadIdx.x & 31;
    const int wave  = threadIdx.x >> 5;
    const int m0    = blockIdx.x * 64;
    const int n0    = blockIdx.y * 128 + wave * 16;
    const int mlane = lane & 15;

    // expert owning this (segment-padded) 64-row block
    int e = NE - 1;
    for (int i = 0; i < NE; ++i) if (m0 >= offs[i]) e = i;

    // ISA 16-bit A layout: lanes<16 hold K[0..7]+[16..23]; lanes>=16 hold K[8..15]+[24..31]
    const int kselA = (lane < 16) ? 0 : 8;
    // ISA 16-bit B layout: lanes<16 hold K[0..7]+[16..23] for N=lane; lanes>=16 hold K[8..15]+[24..31]
    const int kselB = (lane < 16) ? 0 : 8;

    const _Float16* a0 = A + (long)(m0 +  0 + mlane) * K;
    const _Float16* a1 = A + (long)(m0 + 16 + mlane) * K;
    const _Float16* a2 = A + (long)(m0 + 32 + mlane) * K;
    const _Float16* a3 = A + (long)(m0 + 48 + mlane) * K;
    const _Float16* brow = Wt + ((long)e * 512 + n0 + mlane) * K;

    v8f c0 = {}, c1 = {}, c2 = {}, c3 = {};
    union { v16h v; v8h h[2]; } a;

    for (int k0 = 0; k0 < K; k0 += 32) {
        __builtin_prefetch(brow + k0 + 64, 0, 1);          // global_prefetch next B chunk
        union { v16h v; v8h h[2]; } bu;
        bu.h[0] = *(const v8h*)(brow + k0 + kselB);
        bu.h[1] = *(const v8h*)(brow + k0 + kselB + 16);
        const v16h b = bu.v;

        a.h[0] = *(const v8h*)(a0 + k0 + kselA);
        a.h[1] = *(const v8h*)(a0 + k0 + kselA + 16);
        c0 = __builtin_amdgcn_wmma_f32_16x16x32_f16(false, a.v, false, b, (short)0, c0, false, false);

        a.h[0] = *(const v8h*)(a1 + k0 + kselA);
        a.h[1] = *(const v8h*)(a1 + k0 + kselA + 16);
        c1 = __builtin_amdgcn_wmma_f32_16x16x32_f16(false, a.v, false, b, (short)0, c1, false, false);

        a.h[0] = *(const v8h*)(a2 + k0 + kselA);
        a.h[1] = *(const v8h*)(a2 + k0 + kselA + 16);
        c2 = __builtin_amdgcn_wmma_f32_16x16x32_f16(false, a.v, false, b, (short)0, c2, false, false);

        a.h[0] = *(const v8h*)(a3 + k0 + kselA);
        a.h[1] = *(const v8h*)(a3 + k0 + kselA + 16);
        c3 = __builtin_amdgcn_wmma_f32_16x16x32_f16(false, a.v, false, b, (short)0, c3, false, false);
    }

    // C/D layout: VGPR j -> row j (lanes<16) / row j+8 (lanes>=16); col = lane%16
    const int col   = n0 + mlane;
    const float bv  = bias[e * 512 + col];
    const int rbase = m0 + ((lane < 16) ? 0 : 8);
    v8f acc[4] = { c0, c1, c2, c3 };
#pragma unroll
    for (int mi = 0; mi < 4; ++mi) {
#pragma unroll
        for (int j = 0; j < 8; ++j) {
            float v = acc[mi][j] + bv;
            v = v > 0.f ? v : 0.f;
            Out[(long)(rbase + mi * 16 + j) * 512 + col] = (_Float16)v;
        }
    }
}

// ---------------- final layer: dot(512) + sigmoid, scatter to token ----------------
__global__ __launch_bounds__(256) void k_final(
    const _Float16* __restrict__ H2, const float* __restrict__ W3,
    const float* __restrict__ b3, const int* __restrict__ offs,
    const int* __restrict__ rowToken, float* __restrict__ out) {
    const int r    = blockIdx.x * 8 + (threadIdx.x >> 5);
    const int lane = threadIdx.x & 31;
    if (r >= MPAD) return;
    const int t = rowToken[r];
    int e = NE - 1;
    for (int i = 0; i < NE; ++i) if (r >= offs[i]) e = i;

    const _Float16* hr = H2 + (long)r * 512;
    const float* w = W3 + e * 512;
    float s = 0.f;
    for (int k = lane; k < 512; k += 32)
        s += (float)hr[k] * w[k];
    for (int off = 16; off; off >>= 1)
        s += __shfl_xor(s, off, 32);
    if (lane == 0 && t >= 0)
        out[t] = 1.f / (1.f + __expf(-(s + b3[e])));
}

extern "C" void kernel_launch(void* const* d_in, const int* in_sizes, int n_in,
                              void* d_out, int out_size, void* d_ws, size_t ws_size,
                              hipStream_t stream) {
    const float* obs = (const float*)d_in[0];
    const float* act = (const float*)d_in[1];
    const int*   sid = (const int*)d_in[2];
    const float* W1  = (const float*)d_in[3];
    const float* b1  = (const float*)d_in[4];
    const float* W2  = (const float*)d_in[5];
    const float* b2  = (const float*)d_in[6];
    const float* W3  = (const float*)d_in[7];
    const float* b3  = (const float*)d_in[8];
    float* out = (float*)d_out;

    char* ws = (char*)d_ws;
    int* offs       = (int*)(ws + WS_OFFS);
    int* cursor     = (int*)(ws + WS_CURSOR);
    int* rowToken   = (int*)(ws + WS_ROWTOK);
    _Float16* W1t   = (_Float16*)(ws + WS_W1T);
    _Float16* W2t   = (_Float16*)(ws + WS_W2T);
    _Float16* X     = (_Float16*)(ws + WS_X);
    _Float16* H1    = (_Float16*)(ws + WS_H1);
    _Float16* H2    = X;   // alias: X dead after GEMM1

    k_route_count<<<1, 256, 0, stream>>>(sid, offs, cursor, rowToken);
    k_route_scatter<<<B_TOK / 256, 256, 0, stream>>>(sid, cursor, rowToken);
    k_build_x<<<(MPAD * DIN) / 256, 256, 0, stream>>>(obs, act, rowToken, X);
    k_transpose_w<<<(2 * NE * HDIM * DIN) / 256, 256, 0, stream>>>(W1, W2, W1t, W2t);

    dim3 gg(MPAD / 64, 512 / 128);
    k_gemm_relu<<<gg, 256, 0, stream>>>(X,  W1t, b1, offs, H1, DIN);
    k_gemm_relu<<<gg, 256, 0, stream>>>(H1, W2t, b2, offs, H2, HDIM);

    k_final<<<MPAD / 8, 256, 0, stream>>>(H2, W3, b3, offs, rowToken, out);
}
